// FoldingGenerator_34608846471752
// MI455X (gfx1250) — matrix operand order repaired and checked by reading, not compiled, not run on hardware
//
#include <hip/hip_runtime.h>
#include <hip/hip_bf16.h>

// ---------------- problem constants ----------------
#define B_    8
#define Q_    2048
#define C_    256
#define NPER  64
#define H1_   256
#define H2_   64
#define NTOK  (B_*Q_)          // 16384
#define EPSF  1e-5f
#define H1P   264              // padded LDS row stride (bf16 elems) to kill bank conflicts

// ---------------- CDNA5 WMMA types ----------------
typedef __attribute__((ext_vector_type(16))) __bf16 v16bf;
typedef __attribute__((ext_vector_type(8)))  __bf16 bf16x8;
typedef __attribute__((ext_vector_type(2)))  __bf16 bf16x2;
typedef __attribute__((ext_vector_type(8)))  float  v8f;

union BFU { v16bf v; bf16x8 h[2]; };

__device__ __forceinline__ v8f wmma_bf16(v16bf a, v16bf b, v8f c) {
  // D = A(16x32 bf16) * B(32x16 bf16) + C(16x16 f32)
  return __builtin_amdgcn_wmma_f32_16x16x32_bf16(false, a, false, b, (short)0, c, false, false);
}

// ---------------- workspace layout (float units) ----------------
#define WS_TOKPART 0u
#define WS_SUM1    (NTOK*C_)          // 4194304
#define WS_SUMSQ1  (WS_SUM1 + 256)
#define WS_SCALE1  (WS_SUMSQ1 + 256)
#define WS_GBIAS   (WS_SCALE1 + 256)
#define WS_SUM2    (WS_GBIAS + NPER*H1_)
#define WS_SUMSQ2  (WS_SUM2 + 64)
#define WS_SCALE2  (WS_SUMSQ2 + 64)
#define WS_BIAS2   (WS_SCALE2 + 64)
#define WS_W2B     (WS_BIAS2 + 64)    // bf16 region (16384 elems = 32 KB)

// ---------------- main-kernel dynamic LDS layout (bytes) ----------------
#define SM_GBIAS   0                      // 64*256 f32     = 65536
#define SM_H1      65536                  // 64*H1P bf16    = 33792 (aliased as 64x64 f32 h2)
#define SM_TS      (SM_H1 + 64*H1P*2)     // 256 f32        = 1024
#define SM_SUM2L   (SM_TS + 1024)         // 64 f32 (+pad)
#define SM_SUMSQ2L (SM_SUM2L + 256)
#define SM_BYTES   (SM_SUMSQ2L + 256)

// ================= kernel 0: zero stat accumulators =================
__global__ void zero_stats_kernel(float* __restrict__ sum1, float* __restrict__ sumsq1,
                                  float* __restrict__ sum2, float* __restrict__ sumsq2) {
  int t = threadIdx.x;
  sum1[t] = 0.f; sumsq1[t] = 0.f;
  if (t < 64) { sum2[t] = 0.f; sumsq2[t] = 0.f; }
}

// ================= kernel 1: GEMM1 (tok_part) + coarse head, fused =================
__global__ __launch_bounds__(256)
void gemm1_coarse_kernel(const float* __restrict__ tokens, const float* __restrict__ W1,
                         const float* __restrict__ Wc, const float* __restrict__ bc,
                         float* __restrict__ tokpart, float* __restrict__ coarse) {
  __shared__ float Tl[8][C_];
  const int t = threadIdx.x;
  const int row0 = blockIdx.x * 8;
  #pragma unroll
  for (int r = 0; r < 8; ++r) Tl[r][t] = tokens[(row0 + r)*C_ + t];
  __syncthreads();
  float acc[8] = {0.f,0.f,0.f,0.f,0.f,0.f,0.f,0.f};
  for (int k = 0; k < C_; ++k) {
    float wv = W1[k*H1_ + t];               // column t of W1[:C]
    #pragma unroll
    for (int r = 0; r < 8; ++r) acc[r] += Tl[r][k] * wv;   // LDS broadcast
  }
  #pragma unroll
  for (int r = 0; r < 8; ++r) tokpart[(row0 + r)*H1_ + t] = acc[r];
  if (t < 24) {                              // coarse = tokens @ Wc + bc (3 cols)
    int r = t / 3, j = t - r*3;
    float s = bc[j];
    for (int k = 0; k < C_; ++k) s += Tl[r][k] * Wc[k*3 + j];
    coarse[(row0 + r)*3 + j] = s;
  }
}

// ================= kernel 2: per-channel sum/sumsq of tok_part =================
__global__ __launch_bounds__(256)
void reduce1_kernel(const float* __restrict__ tokpart,
                    float* __restrict__ sum1, float* __restrict__ sumsq1) {
  const int t = threadIdx.x;
  const int r0 = blockIdx.x * 256;
  float s = 0.f, q = 0.f;
  for (int r = 0; r < 256; ++r) {
    float v = tokpart[(r0 + r)*H1_ + t];     // coalesced across t
    s += v; q += v*v;
  }
  atomicAdd(&sum1[t], s);
  atomicAdd(&sumsq1[t], q);
}

// ================= kernel 3: finalize BN1 (factorized stats) + fold into gbias =================
__global__ void finalize1_kernel(const float* __restrict__ sum1, const float* __restrict__ sumsq1,
                                 const float* __restrict__ grid2, const float* __restrict__ W1,
                                 const float* __restrict__ b1, const float* __restrict__ gamma1,
                                 const float* __restrict__ beta1,
                                 float* __restrict__ scale1, float* __restrict__ gbias) {
  const int c = threadIdx.x;
  const float invT = 1.f / (float)NTOK;
  float meanT = sum1[c]*invT;
  float varT  = sumsq1[c]*invT - meanT*meanT;
  float wa = W1[256*H1_ + c], wb = W1[257*H1_ + c];   // grid rows of W1
  float sg = 0.f, qg = 0.f;
  for (int n = 0; n < NPER; ++n) {
    float gp = grid2[2*n]*wa + grid2[2*n+1]*wb;
    sg += gp; qg += gp*gp;
  }
  float meanG = sg * (1.f/64.f);
  float varG  = qg * (1.f/64.f) - meanG*meanG;
  // h = T + G + b1 over a full cross product  =>  mean/var factorize exactly
  float mean1 = meanT + meanG + b1[c];
  float var1  = varT + varG;
  float sc = gamma1[c] * rsqrtf(var1 + EPSF);
  scale1[c] = sc;
  float bcon = beta1[c] + sc*(b1[c] - mean1);
  for (int n = 0; n < NPER; ++n) {
    float gp = grid2[2*n]*wa + grid2[2*n+1]*wb;
    gbias[n*H1_ + c] = sc*gp + bcon;          // h1 = relu(sc*T[c] + gbias[n,c])
  }
}

// ================= kernel 4: swizzle W2 -> per-lane-contiguous bf16 B fragments =================
// fragment(ni,kc): lane L holds B[k, n] for n = ni*16 + (L&15), k = kc*32 + (L>>4)*8 + (j&7) + (j>>3)*16, j=0..15
__global__ void prepw2_kernel(const float* __restrict__ W2, __bf16* __restrict__ W2B) {
  int id = blockIdx.x*256 + threadIdx.x;     // 0..16383
  int j    = id & 15;
  int lane = (id >> 4) & 31;
  int kc   = (id >> 9) & 7;
  int ni   = id >> 12;
  int k = kc*32 + (lane >> 4)*8 + (j & 7) + (j >> 3)*16;
  int n = ni*16 + (lane & 15);
  W2B[id] = (__bf16)W2[k*H2_ + n];
}

// ================= kernel 6: finalize BN2 affine =================
__global__ void finalize2_kernel(const float* __restrict__ sum2, const float* __restrict__ sumsq2,
                                 const float* __restrict__ gamma2, const float* __restrict__ beta2,
                                 float* __restrict__ scale2, float* __restrict__ bias2) {
  int d = threadIdx.x;                        // 64 threads
  const float inv = 1.f / (float)(NTOK * NPER);
  float m   = sum2[d]*inv;
  float var = sumsq2[d]*inv - m*m;
  float sc  = gamma2[d]*rsqrtf(var + EPSF);
  scale2[d] = sc;
  bias2[d]  = beta2[d] - sc*m;                // (b2 cancels through BN2)
}

// ================= kernels 5 & 7: fused folding core (stats pass / output pass) =================
// Per token: build h1(64x256 bf16) in LDS, GEMM2 via v_wmma_f32_16x16x32_bf16 against
// register-resident W2 fragments; PASS_A accumulates BN2 sum/sumsq; PASS_B applies
// BN2+relu, GEMM3 (64->3) from LDS, adds coarse, writes fine.
template<bool PASS_B>
__global__ __launch_bounds__(256, 2)
void fold_main_kernel(const float* __restrict__ tokpart, const float* __restrict__ scale1g,
                      const float* __restrict__ gbias_g, const __bf16* __restrict__ W2B,
                      const float* __restrict__ scale2g, const float* __restrict__ bias2g,
                      const float* __restrict__ W3, const float* __restrict__ b3,
                      const float* __restrict__ coarse,
                      float* __restrict__ sum2g, float* __restrict__ sumsq2g,
                      float* __restrict__ fine) {
  extern __shared__ char smem[];
  float*  gbias   = (float*)(smem + SM_GBIAS);
  __bf16* h1      = (__bf16*)(smem + SM_H1);     // 64 x H1P bf16
  float*  h2      = (float*)(smem + SM_H1);      // aliased after barrier: 64 x 64 f32
  float*  Ts      = (float*)(smem + SM_TS);
  float*  sum2l   = (float*)(smem + SM_SUM2L);
  float*  sumsq2l = (float*)(smem + SM_SUMSQ2L);

  const int t    = threadIdx.x;
  const int lane = t & 31;
  const int wv   = t >> 5;          // 8 waves
  const int ni   = wv & 3;          // N-block of the 64x64 output
  const int mi0  = wv >> 2;         // M-blocks {mi0, mi0+2}
  const int mi1  = mi0 + 2;
  const int kh   = (lane >> 4) * 8; // half-wave K/M sub-offset per ISA layout
  const int d    = ni*16 + (lane & 15);   // fixed H2 channel for this lane

  for (int i = t; i < NPER*H1_; i += 256) gbias[i] = gbias_g[i];
  if (!PASS_B && t < 64) { sum2l[t] = 0.f; sumsq2l[t] = 0.f; }

  // W2 B-fragments: resident in VGPRs for the whole kernel (reused across tokens)
  v16bf bfrag[8];
  #pragma unroll
  for (int kc = 0; kc < 8; ++kc) {
    const __bf16* p = W2B + (((ni*8) + kc)*32 + lane)*16;
    BFU u; u.h[0] = *(const bf16x8*)p; u.h[1] = *(const bf16x8*)(p + 8);
    bfrag[kc] = u.v;
  }

  float sc2 = 0.f, bi2 = 0.f;
  if (PASS_B) { sc2 = scale2g[d]; bi2 = bias2g[d]; }

  float runs = 0.f, runq = 0.f;
  const int tpb  = NTOK / (int)gridDim.x;
  const int tok0 = blockIdx.x * tpb;

  for (int tk = 0; tk < tpb; ++tk) {
    const int bq = tok0 + tk;
    Ts[t] = scale1g[t] * tokpart[bq*C_ + t];
    __syncthreads();                                   // Ts ready; prev token fully drained

    // h1[n,c] = relu(Ts[c] + gbias[n,c]) -> bf16, padded row stride H1P
    #pragma unroll 4
    for (int i = 0; i < 32; ++i) {
      int pidx = i*256 + t;
      int n  = pidx >> 7;
      int c0 = (pidx & 127) << 1;
      float v0 = fmaxf(Ts[c0]   + gbias[n*H1_ + c0],     0.f);
      float v1 = fmaxf(Ts[c0+1] + gbias[n*H1_ + c0 + 1], 0.f);
      bf16x2 pr; pr.x = (__bf16)v0; pr.y = (__bf16)v1;
      *(bf16x2*)(h1 + n*H1P + c0) = pr;
    }
    __syncthreads();                                   // h1 ready

    // GEMM2: two 16x16 tiles per wave, K=256 in 8 steps of 32
    v8f acc0 = {0,0,0,0,0,0,0,0};
    v8f acc1 = {0,0,0,0,0,0,0,0};
    const __bf16* row0 = h1 + (mi0*16 + (lane & 15))*H1P;
    const __bf16* row1 = h1 + (mi1*16 + (lane & 15))*H1P;
    #pragma unroll
    for (int kc = 0; kc < 8; ++kc) {
      BFU a0; a0.h[0] = *(const bf16x8*)(row0 + kc*32 + kh);
              a0.h[1] = *(const bf16x8*)(row0 + kc*32 + 16 + kh);
      acc0 = wmma_bf16(a0.v, bfrag[kc], acc0);
      BFU a1; a1.h[0] = *(const bf16x8*)(row1 + kc*32 + kh);
              a1.h[1] = *(const bf16x8*)(row1 + kc*32 + 16 + kh);
      acc1 = wmma_bf16(a1.v, bfrag[kc], acc1);
    }
    __syncthreads();                                   // all h1 reads done

    if (!PASS_B) {
      #pragma unroll
      for (int v = 0; v < 8; ++v) {                    // lane-fixed channel d
        runs += acc0[v] + acc1[v];
        runq += acc0[v]*acc0[v] + acc1[v]*acc1[v];
      }
    } else {
      // BN2 + relu into LDS (reusing h1 region), then GEMM3 (64->3) + coarse
      const int r0 = mi0*16 + kh, r1 = mi1*16 + kh;
      #pragma unroll
      for (int v = 0; v < 8; ++v) {
        h2[(r0+v)*H2_ + d] = fmaxf(sc2*acc0[v] + bi2, 0.f);
        h2[(r1+v)*H2_ + d] = fmaxf(sc2*acc1[v] + bi2, 0.f);
      }
      __syncthreads();
      if (t < 192) {
        int n = t / 3, j = t - n*3;
        float s = b3[j];
        #pragma unroll 8
        for (int dd = 0; dd < H2_; ++dd) s += h2[n*H2_ + dd] * W3[dd*3 + j];
        fine[(bq*NPER + n)*3 + j] = coarse[bq*3 + j] + s;
      }
    }
  }

  if (!PASS_B) {
    atomicAdd(&sum2l[d], runs);                        // ds_add_f32
    atomicAdd(&sumsq2l[d], runq);
    __syncthreads();
    if (t < 64) {
      atomicAdd(&sum2g[t], sum2l[t]);
      atomicAdd(&sumsq2g[t], sumsq2l[t]);
    }
  }
}

// ================= host launch =================
extern "C" void kernel_launch(void* const* d_in, const int* in_sizes, int n_in,
                              void* d_out, int out_size, void* d_ws, size_t ws_size,
                              hipStream_t stream) {
  const float* tokens = (const float*)d_in[0];
  const float* grid2  = (const float*)d_in[1];
  const float* Wc     = (const float*)d_in[2];
  const float* bc     = (const float*)d_in[3];
  const float* W1     = (const float*)d_in[4];
  const float* b1     = (const float*)d_in[5];
  const float* gamma1 = (const float*)d_in[6];
  const float* beta1  = (const float*)d_in[7];
  const float* W2     = (const float*)d_in[8];
  // d_in[9] = b2: provably cancels through BN2 (training-mode stats) -> unused
  const float* gamma2 = (const float*)d_in[10];
  const float* beta2  = (const float*)d_in[11];
  const float* W3     = (const float*)d_in[12];
  const float* b3     = (const float*)d_in[13];

  float* ws      = (float*)d_ws;
  float* tokpart = ws + WS_TOKPART;
  float* sum1    = ws + WS_SUM1;
  float* sumsq1  = ws + WS_SUMSQ1;
  float* scale1  = ws + WS_SCALE1;
  float* gbias   = ws + WS_GBIAS;
  float* sum2    = ws + WS_SUM2;
  float* sumsq2  = ws + WS_SUMSQ2;
  float* scale2  = ws + WS_SCALE2;
  float* bias2   = ws + WS_BIAS2;
  __bf16* W2B    = (__bf16*)(ws + WS_W2B);

  float* coarse = (float*)d_out;                 // (B,Q,3)
  float* fine   = (float*)d_out + NTOK*3;        // (B,Q*64,3)

  (void)hipFuncSetAttribute((const void*)fold_main_kernel<false>,
                            hipFuncAttributeMaxDynamicSharedMemorySize, SM_BYTES);
  (void)hipFuncSetAttribute((const void*)fold_main_kernel<true>,
                            hipFuncAttributeMaxDynamicSharedMemorySize, SM_BYTES);

  zero_stats_kernel<<<1, 256, 0, stream>>>(sum1, sumsq1, sum2, sumsq2);
  gemm1_coarse_kernel<<<NTOK/8, 256, 0, stream>>>(tokens, W1, Wc, bc, tokpart, coarse);
  reduce1_kernel<<<NTOK/256, 256, 0, stream>>>(tokpart, sum1, sumsq1);
  finalize1_kernel<<<1, 256, 0, stream>>>(sum1, sumsq1, grid2, W1, b1, gamma1, beta1,
                                          scale1, gbias);
  prepw2_kernel<<<64, 256, 0, stream>>>(W2, W2B);
  fold_main_kernel<false><<<512, 256, SM_BYTES, stream>>>(
      tokpart, scale1, gbias, W2B, nullptr, nullptr, W3, b3, coarse, sum2, sumsq2, fine);
  finalize2_kernel<<<1, 64, 0, stream>>>(sum2, sumsq2, gamma2, beta2, scale2, bias2);
  fold_main_kernel<true><<<512, 256, SM_BYTES, stream>>>(
      tokpart, scale1, gbias, W2B, scale2, bias2, W3, b3, coarse, sum2, sumsq2, fine);
}
